// QKVMultiheadAttention_80762565034148
// MI455X (gfx1250) — matrix operand order repaired and checked by reading, not compiled, not run on hardware
//
#include <hip/hip_runtime.h>
#include <math.h>

// Problem constants (from reference): B=2, S=2048, D=2048, H=16, HD=128
#define B_  2
#define S_  2048
#define D_  2048
#define H_  16
#define HD_ 128
#define M_  (B_ * S_)      // 4096 total rows
#define QK_SCALE 0.08838834764831845f  // 1/sqrt(128)

typedef __attribute__((ext_vector_type(8)))  _Float16 v8h;
typedef __attribute__((ext_vector_type(16))) _Float16 v16h;
typedef __attribute__((ext_vector_type(8)))  float    v8f;

// Concatenate two 8x f16 (16-byte) loads into one 16x f16 WMMA fragment.
static __device__ __forceinline__ v16h frag_cat(const _Float16* p0, const _Float16* p1) {
    v8h lo = *(const v8h*)p0;
    v8h hi = *(const v8h*)p1;
    return __builtin_shufflevector(lo, hi, 0, 1, 2, 3, 4, 5, 6, 7,
                                           8, 9, 10, 11, 12, 13, 14, 15);
}

// ---------------------------------------------------------------------------
// f32 -> f16 conversion (grid-stride)
// ---------------------------------------------------------------------------
__global__ __launch_bounds__(256)
void cvt_f32_to_f16(const float* __restrict__ in,
                    _Float16* __restrict__ out, int n) {
    int i = blockIdx.x * blockDim.x + threadIdx.x;
    int stride = gridDim.x * blockDim.x;
    for (; i < n; i += stride) out[i] = (_Float16)in[i];
}

// ---------------------------------------------------------------------------
// WMMA GEMM: y[m,n] = sum_k A[m,k] * W[n,k] + bias[n]   (i.e. A @ W.T + b)
// One wave per 32x64 output tile (2 M-subtiles x 4 N-subtiles): every B
// fragment feeds 2 WMMAs, halving W traffic. K-loop double-buffered so the
// next step's 12 b128 loads are in flight during the current 8-WMMA chain.
// MODE 0: Q  -> f16, scaled by 1/sqrt(HD), layout [b,h,s,hd]
// MODE 1: K  -> f16, layout [b,h,s,hd]
// MODE 2: V  -> f16, TRANSPOSED layout [b,h,hd,s]
// MODE 3: out-> f32, layout [m, n] (row-major [B,S,D])
// ---------------------------------------------------------------------------
template <int MODE>
__global__ __launch_bounds__(32)
void gemm32x64_f16wmma(const _Float16* __restrict__ A,
                       const _Float16* __restrict__ W,
                       const float* __restrict__ bias,
                       _Float16* __restrict__ outh,
                       float* __restrict__ outf) {
    const int lane    = threadIdx.x;        // wave32
    const int row0    = blockIdx.x * 32;    // M tile (2 x 16)
    const int col0    = blockIdx.y * 64;    // N strip (4 x 16)
    const int nsub    = lane & 15;
    const bool hihalf = lane >= 16;
    const int kbandA  = hihalf ? 16 : 0;    // 16-bit A frag: lanes 0-15 K=0..15, lanes 16-31 K=16..31
    const int kbB     = hihalf ? 16 : 0;    // 16-bit B frag: lanes 0-15 K=0..15, lanes 16-31 K=16..31
    const int rowbase = hihalf ? 8 : 0;     // f32 C/D frag: VGPR r -> row r (+8 for hi half)

    const _Float16* arow[2];
#pragma unroll
    for (int u = 0; u < 2; ++u)
        arow[u] = A + (size_t)(row0 + u * 16 + nsub) * D_ + kbandA;
    const _Float16* wrow[4];
#pragma unroll
    for (int t = 0; t < 4; ++t)
        wrow[t] = W + (size_t)(col0 + t * 16 + nsub) * D_ + kbB;

    v8f acc[2][4] = {};

    // software pipeline prologue: fragments for k0 = 0
    v16h a_cur[2], b_cur[4];
#pragma unroll
    for (int u = 0; u < 2; ++u) a_cur[u] = frag_cat(arow[u], arow[u] + 8);
#pragma unroll
    for (int t = 0; t < 4; ++t) b_cur[t] = frag_cat(wrow[t], wrow[t] + 8);

    for (int k0 = 0; k0 < D_; k0 += 32) {
        v16h a_nxt[2], b_nxt[4];
        const int k1 = k0 + 32;
        if (k1 < D_) {
            __builtin_prefetch(arow[0] + k1 + 96, 0, 0);
#pragma unroll
            for (int u = 0; u < 2; ++u)
                a_nxt[u] = frag_cat(arow[u] + k1, arow[u] + k1 + 8);
#pragma unroll
            for (int t = 0; t < 4; ++t)
                b_nxt[t] = frag_cat(wrow[t] + k1, wrow[t] + k1 + 8);
        }
#pragma unroll
        for (int t = 0; t < 4; ++t) {
#pragma unroll
            for (int u = 0; u < 2; ++u)
                acc[u][t] = __builtin_amdgcn_wmma_f32_16x16x32_f16(
                    false, a_cur[u], false, b_cur[t], (short)0, acc[u][t],
                    false, false);
        }
#pragma unroll
        for (int u = 0; u < 2; ++u) a_cur[u] = a_nxt[u];
#pragma unroll
        for (int t = 0; t < 4; ++t) b_cur[t] = b_nxt[t];
    }

#pragma unroll
    for (int t = 0; t < 4; ++t) {
        const int n  = col0 + t * 16 + nsub;
        const float bv = bias[n];
#pragma unroll
        for (int u = 0; u < 2; ++u) {
#pragma unroll
            for (int r = 0; r < 8; ++r) {
                const int m = row0 + u * 16 + rowbase + r;
                const float v = acc[u][t][r] + bv;
                if (MODE == 3) {
                    outf[(size_t)m * D_ + n] = v;
                } else {
                    const int b = m >> 11;            // m / S
                    const int s = m & (S_ - 1);
                    const int h = n >> 7;             // n / HD
                    const int d = n & (HD_ - 1);
                    if (MODE == 0)
                        outh[((size_t)(b * H_ + h) * S_ + s) * HD_ + d] =
                            (_Float16)(v * QK_SCALE);
                    else if (MODE == 1)
                        outh[((size_t)(b * H_ + h) * S_ + s) * HD_ + d] =
                            (_Float16)v;
                    else // MODE == 2: V stored transposed [b,h,hd,s]
                        outh[((size_t)(b * H_ + h) * HD_ + d) * S_ + s] =
                            (_Float16)v;
                }
            }
        }
    }
}

// ---------------------------------------------------------------------------
// Flash attention: one wave per 16-query tile per (b,h).
// Per 32-key step:
//   1) preload all 8 K-fragments, then 8 back-to-back score WMMAs
//   2) preload all 8 V-fragments BEFORE the softmax barrier so the global
//      loads fly during the v_exp_f32 chain
//   3) online-softmax in LDS, rescale accumulator, 8 back-to-back PV WMMAs
// __launch_bounds__(32) gives the full VGPR budget (no spills).
// ---------------------------------------------------------------------------
__global__ __launch_bounds__(32)
void attn_flash16(const _Float16* __restrict__ Q,
                  const _Float16* __restrict__ K,
                  const _Float16* __restrict__ Vt,
                  _Float16* __restrict__ Ah) {
    __shared__ __align__(16) float    sS[16 * 32];   // raw score tile
    __shared__ __align__(16) _Float16 sP[16 * 32];   // exp() tile as f16
    __shared__ float sm[16], sl[16], ssc[16];        // row max / sum / rescale

    const int lane    = threadIdx.x;
    const int q0      = blockIdx.x * 16;
    const int bh      = blockIdx.y;                  // b*H + h
    const int nsub    = lane & 15;
    const bool hihalf = lane >= 16;
    const int kbandA  = hihalf ? 16 : 0;
    const int kbB     = hihalf ? 16 : 0;
    const int rowbase = hihalf ? 8 : 0;

    const _Float16* Qb = Q  + ((size_t)bh * S_ + q0) * HD_;
    const _Float16* Kb = K  + (size_t)bh * S_ * HD_;
    const _Float16* Vb = Vt + (size_t)bh * HD_ * S_;

    // Q fragments over the full head dim (K = 4 x 32), loaded once, resident.
    v16h qf[4];
    const _Float16* qrow = Qb + (size_t)nsub * HD_;
#pragma unroll
    for (int kk = 0; kk < 4; ++kk)
        qf[kk] = frag_cat(qrow + kk * 32 + kbandA, qrow + kk * 32 + kbandA + 8);

    if (lane < 16) { sm[lane] = -INFINITY; sl[lane] = 0.0f; }
    __syncthreads();

    v8f acc[8] = {};  // 16 x 128 f32 accumulator (8 column tiles)

    for (int kv0 = 0; kv0 < S_; kv0 += 32) {
        // ---- (1) scores: load ALL 8 K-fragments, then pure WMMA chain
        v16h kf[2][4];
#pragma unroll
        for (int j = 0; j < 2; ++j) {
            const _Float16* krow =
                Kb + (size_t)(kv0 + j * 16 + nsub) * HD_ + kbB;
#pragma unroll
            for (int kk = 0; kk < 4; ++kk)
                kf[j][kk] = frag_cat(krow + kk * 32, krow + kk * 32 + 8);
        }
        v8f sc[2] = {};
#pragma unroll
        for (int kk = 0; kk < 4; ++kk) {
#pragma unroll
            for (int j = 0; j < 2; ++j)
                sc[j] = __builtin_amdgcn_wmma_f32_16x16x32_f16(
                    false, qf[kk], false, kf[j][kk], (short)0, sc[j],
                    false, false);
        }
#pragma unroll
        for (int j = 0; j < 2; ++j)
#pragma unroll
            for (int r = 0; r < 8; ++r)
                sS[(rowbase + r) * 32 + j * 16 + nsub] = sc[j][r];

        // ---- (2) V-fragment preload: independent of softmax, overlaps it
        v16h vf[8];
#pragma unroll
        for (int i = 0; i < 8; ++i) {
            const _Float16* vrow =
                Vb + (size_t)(i * 16 + nsub) * S_ + kv0 + kbB;
            vf[i] = frag_cat(vrow, vrow + 8);
        }
        __syncthreads();

        // ---- (3) online softmax: lanes 0..15 each own one query row
        if (lane < 16) {
            const float mold = sm[lane];
            float mx = mold;
            float rowv[32];
#pragma unroll
            for (int i = 0; i < 32; ++i) {
                rowv[i] = sS[lane * 32 + i];
                mx = fmaxf(mx, rowv[i]);
            }
            const float scale = __expf(mold - mx);
            float lsum = sl[lane] * scale;
#pragma unroll
            for (int i = 0; i < 32; ++i) {
                const float p = __expf(rowv[i] - mx);
                lsum += p;
                sP[lane * 32 + i] = (_Float16)p;
            }
            sm[lane] = mx; sl[lane] = lsum; ssc[lane] = scale;
        }
        __syncthreads();

        // ---- rescale accumulator rows by exp(m_old - m_new)
        float sc8[8];
#pragma unroll
        for (int r = 0; r < 8; ++r) sc8[r] = ssc[rowbase + r];
#pragma unroll
        for (int ct = 0; ct < 8; ++ct)
#pragma unroll
            for (int r = 0; r < 8; ++r) acc[ct][r] *= sc8[r];

        // ---- acc += P(16x32) @ V(32x128): pure WMMA chain (vf resident)
        const _Float16* prow = &sP[nsub * 32];
        v16h pf = frag_cat(prow + kbandA, prow + kbandA + 8);
#pragma unroll
        for (int i = 0; i < 8; ++i)
            acc[i] = __builtin_amdgcn_wmma_f32_16x16x32_f16(
                false, pf, false, vf[i], (short)0, acc[i], false, false);
        __syncthreads();
    }

    // ---- normalize by row sums, store concat-heads f16 [b, s, h*HD + d]
    float linv[8];
#pragma unroll
    for (int r = 0; r < 8; ++r) linv[r] = 1.0f / sl[rowbase + r];
    const int b = bh >> 4, h = bh & 15;
#pragma unroll
    for (int ct = 0; ct < 8; ++ct) {
#pragma unroll
        for (int r = 0; r < 8; ++r) {
            const int srow = q0 + rowbase + r;
            const int col  = h * HD_ + ct * 16 + nsub;
            Ah[((size_t)b * S_ + srow) * D_ + col] =
                (_Float16)(acc[ct][r] * linv[r]);
        }
    }
}

// ---------------------------------------------------------------------------
// Host-side launcher
// ---------------------------------------------------------------------------
extern "C" void kernel_launch(void* const* d_in, const int* in_sizes, int n_in,
                              void* d_out, int out_size, void* d_ws, size_t ws_size,
                              hipStream_t stream) {
    const float* x  = (const float*)d_in[0];
    const float* Wq = (const float*)d_in[1];
    const float* bq = (const float*)d_in[2];
    const float* Wk = (const float*)d_in[3];
    const float* bk = (const float*)d_in[4];
    const float* Wv = (const float*)d_in[5];
    const float* bv = (const float*)d_in[6];
    const float* Wo = (const float*)d_in[7];
    const float* bo = (const float*)d_in[8];
    float* out = (float*)d_out;

    const size_t NX = (size_t)B_ * S_ * D_;  // 8,388,608 elements
    const size_t NW = (size_t)D_ * D_;       // 4,194,304 elements

    _Float16* ws  = (_Float16*)d_ws;
    _Float16* xh  = ws; ws += NX;
    _Float16* Wqh = ws; ws += NW;
    _Float16* Wkh = ws; ws += NW;
    _Float16* Wvh = ws; ws += NW;
    _Float16* Woh = ws; ws += NW;
    _Float16* Qh  = ws; ws += NX;   // [b,h,s,hd], pre-scaled
    _Float16* Kh  = ws; ws += NX;   // [b,h,s,hd]
    _Float16* Vth = ws; ws += NX;   // [b,h,hd,s] (transposed for PV fragments)
    _Float16* Ahh = ws; ws += NX;   // concat-heads attention output

    // 1) downconvert activations + weights to f16
    cvt_f32_to_f16<<<2048, 256, 0, stream>>>(x,  xh,  (int)NX);
    cvt_f32_to_f16<<<1024, 256, 0, stream>>>(Wq, Wqh, (int)NW);
    cvt_f32_to_f16<<<1024, 256, 0, stream>>>(Wk, Wkh, (int)NW);
    cvt_f32_to_f16<<<1024, 256, 0, stream>>>(Wv, Wvh, (int)NW);
    cvt_f32_to_f16<<<1024, 256, 0, stream>>>(Wo, Woh, (int)NW);

    // 2) QKV projections (WMMA GEMMs), 32x64 tile per wave
    dim3 gg(M_ / 32, D_ / 64);
    gemm32x64_f16wmma<0><<<gg, 32, 0, stream>>>(xh, Wqh, bq, Qh,  nullptr);
    gemm32x64_f16wmma<1><<<gg, 32, 0, stream>>>(xh, Wkh, bk, Kh,  nullptr);
    gemm32x64_f16wmma<2><<<gg, 32, 0, stream>>>(xh, Wvh, bv, Vth, nullptr);

    // 3) flash attention per (b,h, 16-query tile)
    attn_flash16<<<dim3(S_ / 16, B_ * H_), 32, 0, stream>>>(Qh, Kh, Vth, Ahh);

    // 4) output projection -> f32
    gemm32x64_f16wmma<3><<<gg, 32, 0, stream>>>(Ahh, Woh, bo, nullptr, out);
}
